// Net_69114613730436
// MI455X (gfx1250) — matrix operand order repaired and checked by reading, not compiled, not run on hardware
//
#include <hip/hip_runtime.h>

// ---------------------------------------------------------------------------
// Problem dimensions (fixed by the reference)
// ---------------------------------------------------------------------------
#define T_LEN 512      // sentence length
#define E_DIM 512      // embedding size
#define H_DIM 1024     // hidden size
#define G_DIM 4096     // 4*H gate rows
#define OUT_N 128
#define NWG   8        // workgroups cooperating per direction in the scan

// GEMM block tiling
#define BM 32
#define BN 256
#define BK 32

typedef __attribute__((ext_vector_type(2)))  _Float16 h2;
typedef __attribute__((ext_vector_type(8)))  _Float16 v8h;
typedef __attribute__((ext_vector_type(16))) _Float16 v16h;
typedef __attribute__((ext_vector_type(8)))  float    v8f;

// pointee types matching the async-to-LDS builtin signatures exactly
typedef int vi2_ __attribute__((vector_size(2 * sizeof(int))));
typedef int vi4_ __attribute__((vector_size(4 * sizeof(int))));

__device__ __forceinline__ float sigmoidf_(float x) { return 1.0f / (1.0f + expf(-x)); }

// ---------------------------------------------------------------------------
// CDNA5 async global->LDS copies (ASYNCcnt-tracked). Guarded so the file
// compiles on toolchains lacking the builtins (falls back to load+ds_store).
// Builtin shape (from clang diagnostic):
//   void __builtin_amdgcn_global_load_async_to_lds_b64 (v2i AS1*, v2i AS3*, Ii, Ii)
//   void __builtin_amdgcn_global_load_async_to_lds_b128(v4i AS1*, v4i AS3*, Ii, Ii)
// ---------------------------------------------------------------------------
#if defined(__has_builtin)
#  if __has_builtin(__builtin_amdgcn_global_load_async_to_lds_b64) && \
      __has_builtin(__builtin_amdgcn_global_load_async_to_lds_b128)
#    define USE_ASYNC_LDS 1
#  endif
#endif

#ifdef USE_ASYNC_LDS
#  define ASYNC_CP64(dst, src)                                        \
      __builtin_amdgcn_global_load_async_to_lds_b64(                  \
          (__attribute__((address_space(1))) vi2_*)(src),             \
          (__attribute__((address_space(3))) vi2_*)(dst), 0, 0)
#  define ASYNC_CP128(dst, src)                                       \
      __builtin_amdgcn_global_load_async_to_lds_b128(                 \
          (__attribute__((address_space(1))) vi4_*)(src),             \
          (__attribute__((address_space(3))) vi4_*)(dst), 0, 0)
#  if __has_builtin(__builtin_amdgcn_s_wait_asynccnt)
#    define WAIT_ASYNC(n) __builtin_amdgcn_s_wait_asynccnt(n)
#  else
#    define WAIT_ASYNC(n) asm volatile("s_wait_asynccnt %0" :: "i"(n) : "memory")
#  endif
#else
   // synchronous fallback: vector global load + ds_store
#  define ASYNC_CP64(dst, src)  (*(uint2*)(dst) = *(const uint2*)(src))
#  define ASYNC_CP128(dst, src) (*(uint4*)(dst) = *(const uint4*)(src))
#  define WAIT_ASYNC(n) ((void)0)
#endif

// ---------------------------------------------------------------------------
// init: zero the ping-pong h vectors and the inter-WGP barrier state.
// ---------------------------------------------------------------------------
__global__ void init_kernel(float* __restrict__ hvec, int* __restrict__ bar) {
    int i = blockIdx.x * blockDim.x + threadIdx.x;
    if (i < 2 * 2 * H_DIM) hvec[i] = 0.0f;
    if (i < 4) bar[i] = 0;
}

// ---------------------------------------------------------------------------
// embedding gather + fp32 -> fp16 convert (forward and reversed sequences)
// ---------------------------------------------------------------------------
__global__ void gather_kernel(const int* __restrict__ x,
                              const float* __restrict__ embed,
                              _Float16* __restrict__ emb16 /* [2][T][E] */) {
    const int t = blockIdx.x;
    const float* rowf = embed + (size_t)x[t] * E_DIM;
    const float* rowb = embed + (size_t)x[T_LEN - 1 - t] * E_DIM;
    for (int k = threadIdx.x; k < E_DIM; k += blockDim.x) {
        emb16[(size_t)t * E_DIM + k]           = (_Float16)rowf[k];
        emb16[(size_t)(T_LEN + t) * E_DIM + k] = (_Float16)rowb[k];
    }
}

__global__ void cvt16_kernel(const float* __restrict__ src,
                             _Float16* __restrict__ dst, int n) {
    int i = blockIdx.x * blockDim.x + threadIdx.x;
    int stride = gridDim.x * blockDim.x;
    for (; i < n; i += stride) dst[i] = (_Float16)src[i];
}

// ---------------------------------------------------------------------------
// Stage one K-step (BMxBK A panel + BNxBK B panel) into LDS via async copies.
// 256 threads: A = 32 rows x 64B (b64 each), B = 256 rows x 64B (4 x b128).
// 5 ASYNCcnt increments per thread per stage.
// ---------------------------------------------------------------------------
__device__ __forceinline__ void stage_tile(const _Float16* __restrict__ A,
                                           const _Float16* __restrict__ B,
                                           int bm0, int bn0, int k0,
                                           _Float16* Al, _Float16* Bl, int tid) {
    {   // A panel
        const int r = tid >> 3;
        const int c = (tid & 7) * 4;          // half index (8 bytes)
        ASYNC_CP64(Al + r * BK + c, A + (size_t)(bm0 + r) * E_DIM + k0 + c);
    }
    {   // B panel: one 64B row per thread
        const _Float16* src = B + (size_t)(bn0 + tid) * E_DIM + k0;
        _Float16* dst = Bl + tid * BK;
#pragma unroll
        for (int i = 0; i < 4; ++i)
            ASYNC_CP128(dst + i * 8, src + i * 8);
    }
}

// ---------------------------------------------------------------------------
// x_gates = emb @ Wih^T + (bih + bhh), WMMA f16 with double-buffered LDS
// staging. Block = 8 waves as 2(M) x 4(N) -> 32x256 C tile; wave = 16x64.
// Fragment layouts used here (contiguous K; A and B match):
//   A (16x32): lane m=l%16; 16 contiguous K (+16 for lanes>=16)
//   B (32x16): lane n=l%16; 16 contiguous K (+16 for lanes>=16)
// ---------------------------------------------------------------------------
__global__ __launch_bounds__(256) void xgates_wmma_kernel(
    const _Float16* __restrict__ emb16,   // [2][T][E]
    const _Float16* __restrict__ wih16,   // [2][G][E]
    const float* __restrict__ bih_f, const float* __restrict__ bhh_f,
    const float* __restrict__ bih_b, const float* __restrict__ bhh_b,
    float* __restrict__ xg)               // [2][T][G]
{
    __shared__ alignas(64) _Float16 Abuf[2][BM * BK];   //  4 KB
    __shared__ alignas(64) _Float16 Bbuf[2][BN * BK];   // 32 KB

    const int tid  = threadIdx.x;
    const int lane = tid & 31;
    const int wv   = tid >> 5;       // 0..7
    const int mt   = wv >> 2;        // 0..1  (M sub-tile)
    const int ng   = wv & 3;         // 0..3  (N group of 64)

    const int blocks_per_dir = (T_LEN / BM) * (G_DIM / BN);   // 16*16 = 256
    const int dir = blockIdx.x / blocks_per_dir;
    const int rem = blockIdx.x % blocks_per_dir;
    const int bm0 = (rem / (G_DIM / BN)) * BM;
    const int bn0 = (rem % (G_DIM / BN)) * BN;

    const _Float16* A = emb16 + (size_t)dir * T_LEN * E_DIM;
    const _Float16* B = wih16 + (size_t)dir * G_DIM * E_DIM;
    const float* bih = dir ? bih_b : bih_f;
    const float* bhh = dir ? bhh_b : bhh_f;

    const int aoff = (lane >= 16) ? 16 : 0;   // A lane-half K offset
    const int boff = (lane >= 16) ? 16 : 0;   // B lane-half K offset
    const int mrow_l = mt * 16 + (lane & 15); // local A row

    v8f acc[4] = {};

    stage_tile(A, B, bm0, bn0, 0, Abuf[0], Bbuf[0], tid);
    int buf = 0;

    for (int k0 = 0; k0 < E_DIM; k0 += BK) {
        if (k0 + BK < E_DIM) {
            stage_tile(A, B, bm0, bn0, k0 + BK, Abuf[buf ^ 1], Bbuf[buf ^ 1], tid);
            WAIT_ASYNC(5);   // retire previous stage; the 5 just-issued may fly
        } else {
            WAIT_ASYNC(0);
        }
        __syncthreads();

        // A fragment from LDS: two 16B chunks per lane
        const _Float16* arow = &Abuf[buf][mrow_l * BK + aoff];
        v8h alo = *(const v8h*)(arow);
        v8h ahi = *(const v8h*)(arow + 8);
        v16h a;
#pragma unroll
        for (int i = 0; i < 8; ++i) { a[i] = alo[i]; a[8 + i] = ahi[i]; }

#pragma unroll
        for (int u = 0; u < 4; ++u) {
            const int nrow_l = ng * 64 + u * 16 + (lane & 15);
            v16h b = *(const v16h*)(&Bbuf[buf][nrow_l * BK + boff]);
            acc[u] = __builtin_amdgcn_wmma_f32_16x16x32_f16(
                false, a, false, b, (short)0, acc[u], false, false);
        }
        __syncthreads();   // protect buf before it is restaged
        buf ^= 1;
    }

    float* xgd = xg + (size_t)dir * T_LEN * G_DIM;
#pragma unroll
    for (int u = 0; u < 4; ++u) {
        const int ncol = bn0 + ng * 64 + u * 16 + (lane & 15);
        const float bias = bih[ncol] + bhh[ncol];
#pragma unroll
        for (int r = 0; r < 8; ++r) {
            const int m = bm0 + mt * 16 + r + ((lane >= 16) ? 8 : 0);
            xgd[(size_t)m * G_DIM + ncol] = acc[u][r] + bias;
        }
    }
}

// ---------------------------------------------------------------------------
// LSTM scan: 2 directions x NWG workgroups, 512 threads each.
// Thread owns gate row n; 1024-long f16xf32 dot against h staged in LDS.
// h ping-pong buffered in global (L2-resident); one device-scope arrive/spin
// barrier per step across the NWG blocks of a direction.
// ---------------------------------------------------------------------------
__global__ __launch_bounds__(512) void lstm_scan_kernel(
    const float* __restrict__ xg,        // [2][T][G]
    const _Float16* __restrict__ whh16,  // [2][G][H]
    float* __restrict__ hs,              // [2][T][H]
    float* __restrict__ hvec,            // [2][2][H]
    int* __restrict__ bar)               // [4]: cnt[2], phase[2]
{
    const int dir = blockIdx.x / NWG;
    const int w   = blockIdx.x % NWG;
    const int tid = threadIdx.x;
    const int gate = tid >> 7;
    const int jj   = tid & 127;
    const int n    = gate * H_DIM + w * 128 + jj;

    __shared__ float hbuf[H_DIM];
    __shared__ float garr[512];
    __shared__ float cbuf[128];
    if (tid < 128) cbuf[tid] = 0.0f;

    const h2*    wrow  = (const h2*)(whh16 + ((size_t)dir * G_DIM + n) * H_DIM);
    const float* xrow  = xg + (size_t)dir * T_LEN * G_DIM;
    float*       hsrow = hs + (size_t)dir * T_LEN * H_DIM;
    float*       hv    = hvec + (size_t)dir * 2 * H_DIM;
    int* cnt   = bar + dir;
    int* phase = bar + 2 + dir;

    for (int t = 0; t < T_LEN; ++t) {
        const int cur = t & 1, nxt = cur ^ 1;

        hbuf[tid] = __hip_atomic_load(&hv[cur * H_DIM + tid],
                                      __ATOMIC_RELAXED, __HIP_MEMORY_SCOPE_AGENT);
        hbuf[512 + tid] = __hip_atomic_load(&hv[cur * H_DIM + 512 + tid],
                                            __ATOMIC_RELAXED, __HIP_MEMORY_SCOPE_AGENT);
        __syncthreads();

        if (t + 1 < T_LEN)
            __builtin_prefetch(&xrow[(size_t)(t + 1) * G_DIM + n], 0, 3);

        float acc = xrow[(size_t)t * G_DIM + n];
#pragma unroll 8
        for (int kk = 0; kk < H_DIM / 2; ++kk) {
            h2 wv = wrow[kk];
            acc = fmaf((float)wv.x, hbuf[2 * kk],     acc);
            acc = fmaf((float)wv.y, hbuf[2 * kk + 1], acc);
        }
        garr[tid] = acc;
        __syncthreads();

        if (tid < 128) {
            const float gi = garr[tid];
            const float gf = garr[128 + tid];
            const float gc = garr[256 + tid];
            const float go = garr[384 + tid];
            float c = sigmoidf_(gf) * cbuf[tid] + sigmoidf_(gi) * tanhf(gc);
            cbuf[tid] = c;
            const float h = sigmoidf_(go) * tanhf(c);
            const int j = w * 128 + tid;
            __hip_atomic_store(&hv[nxt * H_DIM + j], h,
                               __ATOMIC_RELAXED, __HIP_MEMORY_SCOPE_AGENT);
            hsrow[(size_t)t * H_DIM + j] = h;
        }
        __syncthreads();

        if (tid == 0) {
            __threadfence();
            int prev = __hip_atomic_fetch_add(cnt, 1, __ATOMIC_ACQ_REL,
                                              __HIP_MEMORY_SCOPE_AGENT);
            if (prev == NWG - 1) {
                __hip_atomic_store(cnt, 0, __ATOMIC_RELAXED, __HIP_MEMORY_SCOPE_AGENT);
                __hip_atomic_store(phase, t + 1, __ATOMIC_RELEASE, __HIP_MEMORY_SCOPE_AGENT);
            } else {
                while (__hip_atomic_load(phase, __ATOMIC_ACQUIRE,
                                         __HIP_MEMORY_SCOPE_AGENT) <= t)
                    __builtin_amdgcn_s_sleep(1);
            }
        }
        __syncthreads();
    }
}

// ---------------------------------------------------------------------------
// head: conv 2x2/s2 + global max pool + fc0 (2560->1024) + fc1 (1024->128)
// ---------------------------------------------------------------------------
__global__ __launch_bounds__(1024) void head_kernel(
    const float* __restrict__ hs,        // [2][T][H]
    const float* __restrict__ conv_w, const float* __restrict__ conv_b,
    const float* __restrict__ fc0_w, const float* __restrict__ fc0_b,
    const float* __restrict__ fc1_w, const float* __restrict__ fc1_b,
    float* __restrict__ out)
{
    __shared__ float comb[2 * H_DIM + H_DIM / 2];
    __shared__ float mid[H_DIM];
    const int tid = threadIdx.x;
    const float* fhs = hs;
    const float* bhs = hs + (size_t)T_LEN * H_DIM;

    comb[tid]        = fhs[(size_t)(T_LEN - 1) * H_DIM + tid];   // f_last
    comb[1536 + tid] = bhs[(size_t)(T_LEN - 1) * H_DIM + tid];   // b_last

    if (tid < H_DIM / 2) {
        const float w00 = conv_w[0], w01 = conv_w[1];
        const float w10 = conv_w[2], w11 = conv_w[3];
        const float cb  = conv_b[0];
        const int cc = 4 * tid;                    // code cols 4p..4p+3
        const float* base = (cc < H_DIM) ? fhs : bhs;
        const int col = (cc < H_DIM) ? cc : cc - H_DIM;
        float m = cb;                              // zero rows give conv_b
        for (int r = 0; r < T_LEN / 2; ++r) {
            const float* r0 = base + (size_t)(2 * r) * H_DIM + col;
            const float* r1 = base + (size_t)(2 * r + 1) * H_DIM + col;
            const float v0 = w00 * r0[0] + w01 * r0[1] + w10 * r1[0] + w11 * r1[1] + cb;
            const float v1 = w00 * r0[2] + w01 * r0[3] + w10 * r1[2] + w11 * r1[3] + cb;
            m = fmaxf(m, fmaxf(v0, v1));
        }
        comb[H_DIM + tid] = m;
    }
    __syncthreads();

    {
        const float* wr = fc0_w + (size_t)tid * 2560;
        float acc = fc0_b[tid];
        for (int k = 0; k < 2560; ++k) acc = fmaf(wr[k], comb[k], acc);
        mid[tid] = acc;
    }
    __syncthreads();

    if (tid < OUT_N) {
        const float* wr = fc1_w + (size_t)tid * H_DIM;
        float acc = fc1_b[tid];
        for (int k = 0; k < H_DIM; ++k) acc = fmaf(wr[k], mid[k], acc);
        out[tid] = acc;
    }
}

// ---------------------------------------------------------------------------
// launch
// ---------------------------------------------------------------------------
extern "C" void kernel_launch(void* const* d_in, const int* in_sizes, int n_in,
                              void* d_out, int out_size, void* d_ws, size_t ws_size,
                              hipStream_t stream) {
    const int*   x      = (const int*)  d_in[0];
    const float* embed  = (const float*)d_in[1];
    const float* Wih_f  = (const float*)d_in[2];
    const float* Whh_f  = (const float*)d_in[3];
    const float* bih_f  = (const float*)d_in[4];
    const float* bhh_f  = (const float*)d_in[5];
    const float* Wih_b  = (const float*)d_in[6];
    const float* Whh_b  = (const float*)d_in[7];
    const float* bih_b  = (const float*)d_in[8];
    const float* bhh_b  = (const float*)d_in[9];
    const float* conv_w = (const float*)d_in[10];
    const float* conv_b = (const float*)d_in[11];
    const float* fc0_w  = (const float*)d_in[12];
    const float* fc0_b  = (const float*)d_in[13];
    const float* fc1_w  = (const float*)d_in[14];
    const float* fc1_b  = (const float*)d_in[15];
    float* out = (float*)d_out;

    char* ws = (char*)d_ws;
    size_t off = 0;
    auto take = [&](size_t bytes) { char* p = ws + off;
                                    off = (off + bytes + 255) & ~(size_t)255; return p; };
    _Float16* emb16 = (_Float16*)take((size_t)2 * T_LEN * E_DIM * 2);
    _Float16* wih16 = (_Float16*)take((size_t)2 * G_DIM * E_DIM * 2);
    _Float16* whh16 = (_Float16*)take((size_t)2 * G_DIM * H_DIM * 2);
    float*    xgbuf = (float*)   take((size_t)2 * T_LEN * G_DIM * 4);
    float*    hsbuf = (float*)   take((size_t)2 * T_LEN * H_DIM * 4);
    float*    hvec  = (float*)   take((size_t)2 * 2 * H_DIM * 4);
    int*      bar   = (int*)     take(256);
    (void)ws_size; (void)in_sizes; (void)n_in; (void)out_size;

    init_kernel<<<16, 256, 0, stream>>>(hvec, bar);
    gather_kernel<<<T_LEN, 256, 0, stream>>>(x, embed, emb16);

    cvt16_kernel<<<1024, 256, 0, stream>>>(Wih_f, wih16,                          G_DIM * E_DIM);
    cvt16_kernel<<<1024, 256, 0, stream>>>(Wih_b, wih16 + (size_t)G_DIM * E_DIM, G_DIM * E_DIM);
    cvt16_kernel<<<1024, 256, 0, stream>>>(Whh_f, whh16,                          G_DIM * H_DIM);
    cvt16_kernel<<<1024, 256, 0, stream>>>(Whh_b, whh16 + (size_t)G_DIM * H_DIM, G_DIM * H_DIM);

    // 2 dirs * 16 m-tiles * 16 n-tiles = 512 blocks
    xgates_wmma_kernel<<<512, 256, 0, stream>>>(emb16, wih16,
                                                bih_f, bhh_f, bih_b, bhh_b, xgbuf);

    lstm_scan_kernel<<<2 * NWG, 512, 0, stream>>>(xgbuf, whh16, hsbuf, hvec, bar);

    head_kernel<<<1, 1024, 0, stream>>>(hsbuf, conv_w, conv_b,
                                        fc0_w, fc0_b, fc1_w, fc1_b, out);
}
